// MultiHeadSelfAttention_69483980915221
// MI455X (gfx1250) — matrix operand order repaired and checked by reading, not compiled, not run on hardware
//
#include <hip/hip_runtime.h>
#include <hip/hip_bf16.h>

// ---------------------------------------------------------------------------
// Fused masked MHA forward for MI455X (gfx1250, wave32, WMMA bf16 16x16x32).
//   B=16, T=1024, D=512, H=8, DH=64.
// Pipeline: prep_weights (W -> W^T bf16) ; ln_kernel (x -> xn bf16) ;
//           proj_kernel (Q,K [B,H,T,64] bf16, V^T [B,H,64,T] bf16) ;
//           attn_kernel (flash-style, transposed S^T/O^T formulation,
//                        K/V chunks staged via global_load_async_to_lds_b128,
//                        double-buffered, shared by all 8 waves of a block).
// Workspace: 3*512*512 + 4*16384*512 bf16 elements ~= 65.5 MB.
// ---------------------------------------------------------------------------

typedef __bf16 bf16;
typedef __attribute__((ext_vector_type(16))) __bf16 v16bf;
typedef __attribute__((ext_vector_type(8)))  __bf16 v8bf;
typedef __attribute__((ext_vector_type(2)))  __bf16 v2bf;
typedef __attribute__((ext_vector_type(8)))  float  v8f;
typedef __attribute__((ext_vector_type(4)))  float  v4f;

#define T_LEN   1024
#define D_DIM   512
#define H_NUM   8
#define DH_DIM  64
#define NEG_BIG 1.0e10f
#define SM_SCALE 0.125f   // 1/sqrt(64)

__device__ __forceinline__ v8f zero8() {
  v8f z = {0.f, 0.f, 0.f, 0.f, 0.f, 0.f, 0.f, 0.f};
  return z;
}

__device__ __forceinline__ v8bf ld8(const bf16* p) { return *(const v8bf*)p; }

// Concatenate two 8-element bf16 vectors (lo -> elements 0..7, hi -> 8..15).
__device__ __forceinline__ v16bf cat(v8bf lo, v8bf hi) {
  v16bf r;
#pragma unroll
  for (int i = 0; i < 8; ++i) { r[i] = lo[i]; r[i + 8] = hi[i]; }
  return r;
}

__device__ __forceinline__ v8f wmma_bf16(v16bf a, v16bf b, v8f c) {
  // D = A(16x32 bf16) * B(32x16 bf16) + C(16x16 f32)
  return __builtin_amdgcn_wmma_f32_16x16x32_bf16(
      /*neg_a=*/false, a, /*neg_b=*/false, b,
      /*c_mod=*/(short)0, c, /*reuse_a=*/false, /*reuse_b=*/false);
}

// Async DMA of 16 bytes global -> LDS (ASYNCcnt-tracked, no VGPR staging).
// VDST VGPR carries the LDS byte offset; VADDR the 64-bit global address.
__device__ __forceinline__ void async_ld16(unsigned lds_off, const bf16* gptr) {
  asm volatile("global_load_async_to_lds_b128 %0, %1, off"
               :: "v"(lds_off), "v"((unsigned long long)(size_t)gptr)
               : "memory");
}

// ---------------------------------------------------------------------------
// Kernel 1: convert + transpose weights.  W[in][out] f32 -> Wt[out][in] bf16.
// ---------------------------------------------------------------------------
__global__ __launch_bounds__(256) void prep_weights(
    const float* __restrict__ wq, const float* __restrict__ wk,
    const float* __restrict__ wv, bf16* __restrict__ wt) {
  int idx = blockIdx.x * blockDim.x + threadIdx.x;   // 0 .. 3*512*512-1
  int m   = idx >> 18;                               // which matrix
  int rem = idx & (D_DIM * D_DIM - 1);
  int n   = rem >> 9;                                // out col
  int k   = rem & (D_DIM - 1);                       // in row
  const float* w = (m == 0) ? wq : (m == 1) ? wk : wv;
  wt[idx] = (bf16)w[k * D_DIM + n];
}

// ---------------------------------------------------------------------------
// Kernel 2: LayerNorm, one wave32 per row of 512, output bf16.
// ---------------------------------------------------------------------------
__global__ __launch_bounds__(256) void ln_kernel(
    const float* __restrict__ x, const float* __restrict__ g,
    const float* __restrict__ bb, bf16* __restrict__ xn) {
  const int lane = threadIdx.x & 31;
  const int wave = threadIdx.x >> 5;
  const int row  = blockIdx.x * 8 + wave;
  const float* xr = x + (size_t)row * D_DIM;
  const int base = lane * 16;

  float e[16];
  v4f t0 = *(const v4f*)(xr + base + 0);
  v4f t1 = *(const v4f*)(xr + base + 4);
  v4f t2 = *(const v4f*)(xr + base + 8);
  v4f t3 = *(const v4f*)(xr + base + 12);
#pragma unroll
  for (int i = 0; i < 4; ++i) {
    e[i] = t0[i]; e[4 + i] = t1[i]; e[8 + i] = t2[i]; e[12 + i] = t3[i];
  }
  float s = 0.f, s2 = 0.f;
#pragma unroll
  for (int i = 0; i < 16; ++i) { s += e[i]; s2 += e[i] * e[i]; }
#pragma unroll
  for (int o = 16; o > 0; o >>= 1) {
    s  += __shfl_xor(s, o);
    s2 += __shfl_xor(s2, o);
  }
  const float mu   = s * (1.0f / 512.0f);
  const float var  = s2 * (1.0f / 512.0f) - mu * mu;
  const float rstd = rsqrtf(var + 1e-5f);

  v8bf o0, o1;
#pragma unroll
  for (int i = 0; i < 8; ++i) {
    o0[i] = (bf16)((e[i] - mu) * rstd * g[base + i] + bb[base + i]);
    o1[i] = (bf16)((e[8 + i] - mu) * rstd * g[base + 8 + i] + bb[base + 8 + i]);
  }
  bf16* dst = xn + (size_t)row * D_DIM + base;
  *(v8bf*)(dst + 0) = o0;
  *(v8bf*)(dst + 8) = o1;
}

// ---------------------------------------------------------------------------
// Kernel 3: projection GEMM.  C[16384,512] = xn(bf16) @ W(bf16) + bias, one
// 16x16 tile per wave, K=512 in 16 chained WMMA(K=32) steps.
//   which=0 -> Q [B,H,T,64]; which=1 -> K [B,H,T,64]; which=2 -> V^T [B,H,64,T]
// ---------------------------------------------------------------------------
__global__ __launch_bounds__(128) void proj_kernel(
    const bf16* __restrict__ xn, const bf16* __restrict__ wt_all,
    const float* __restrict__ bq, const float* __restrict__ bk,
    const float* __restrict__ bv, bf16* __restrict__ qout,
    bf16* __restrict__ kout, bf16* __restrict__ vtout) {
  const int lane   = threadIdx.x & 31;
  const int wave   = threadIdx.x >> 5;
  const int rt     = blockIdx.x * 4 + wave;   // row tile (0..1023)
  const int ct     = blockIdx.y;              // col tile (0..31)
  const int which  = blockIdx.z;              // 0=Q 1=K 2=V
  const int n      = lane & 15;
  const int lh     = lane >> 4;               // lane half
  const int klo    = lh * 8;                  // A-layout K interleave base
  const int row0   = rt * 16;
  const int c      = ct * 16 + n;             // this lane's output column

  const bf16* wt   = wt_all + (size_t)which * D_DIM * D_DIM;
  const float* bias = (which == 0) ? bq : (which == 1) ? bk : bv;

  // A: xn row for M = n (lanes 0-15 and 16-31 both map M = lane%16).
  const bf16* arow = xn + (size_t)(row0 + n) * D_DIM;
  // B: Wt row for N = c ; table-order K interleave per lane half, same as A.
  const bf16* brow = wt + (size_t)c * D_DIM;

  v8f acc = zero8();
#pragma unroll 4
  for (int kk = 0; kk < D_DIM; kk += 32) {
    v16bf a = cat(ld8(arow + kk + klo), ld8(arow + kk + klo + 16));
    v16bf b = cat(ld8(brow + kk + lh * 8), ld8(brow + kk + lh * 8 + 16));
    acc = wmma_bf16(a, b, acc);
  }

  const float bc  = bias[c];
  const int   b_  = row0 >> 10;       // batch
  const int   t0  = row0 & (T_LEN - 1);
  const int   h   = c >> 6;
  const int   dh  = c & (DH_DIM - 1);
  const int   off = lh * 8;           // C/D layout: VGPR r -> M = r + off

  if (which < 2) {
    bf16* dst = (which == 0 ? qout : kout);
#pragma unroll
    for (int r = 0; r < 8; ++r) {
      const int t = t0 + off + r;
      dst[((size_t)(b_ * H_NUM + h) * T_LEN + t) * DH_DIM + dh] =
          (bf16)(acc[r] + bc);
    }
  } else {
    // V transposed: vt[(bh*64 + dh)*T + t]; consecutive r -> consecutive t.
    bf16* dst = vtout + ((size_t)(b_ * H_NUM + h) * DH_DIM + dh) * T_LEN +
                t0 + off;
#pragma unroll
    for (int r = 0; r < 8; r += 2) {
      v2bf p = { (bf16)(acc[r] + bc), (bf16)(acc[r + 1] + bc) };
      *(v2bf*)(dst + r) = p;
    }
  }
}

// ---------------------------------------------------------------------------
// Kernel 4: flash attention (transposed form).  One wave per 16-query tile.
//   S^T = K * Q^T  (keys on M/VGPRs, queries on N/lanes)  -> softmax over
//   keys is within-lane across accumulator VGPRs + one shfl_xor(16) merge.
//   O^T = V^T * P^T ; P^T goes through a per-wave LDS tile to reach B-layout.
//   K/V chunks (32 keys) are DMAed once per block into double-buffered LDS
//   with global_load_async_to_lds_b128 and shared by all 8 waves.
// ---------------------------------------------------------------------------
__global__ __launch_bounds__(256, 2) void attn_kernel(
    const bf16* __restrict__ q, const bf16* __restrict__ k,
    const bf16* __restrict__ vt, const int* __restrict__ mask,
    float* __restrict__ out) {
  __shared__ float smadd[T_LEN];           // (1-m)*1e10 (pre-softmax subtract)
  __shared__ float smmul[T_LEN];           // m           (post-softmax mult)
  __shared__ bf16  spb[8][16 * 32];        // per-wave P^T staging tile
  __shared__ bf16  skb[2][32 * DH_DIM];    // K chunk: [key 0..31][dh 0..63]
  __shared__ bf16  svb[2][DH_DIM * 32];    // V^T chunk: [dh 0..63][key 0..31]

  const int tid  = threadIdx.x;
  const int lane = tid & 31;
  const int wave = tid >> 5;
  const int bh   = blockIdx.y;             // (b,h) pair, 0..127
  const int b_   = bh >> 3;
  const int h    = bh & 7;
  const int qt   = blockIdx.x * 8 + wave;  // query tile, 0..63

  const bf16* kbase = k  + (size_t)bh * T_LEN * DH_DIM;
  const bf16* vbase = vt + (size_t)bh * DH_DIM * T_LEN;

  // --- cooperative async stage of one 32-key chunk (8KB: 2 b128 per thread)
  const int sdh   = tid >> 2;              // V row (dh) for this thread
  const int spc   = tid & 3;               // 16-byte piece within the V row
  const unsigned koff0 = (unsigned)(size_t)(&skb[0][0]);
  const unsigned koff1 = (unsigned)(size_t)(&skb[1][0]);
  const unsigned voff0 = (unsigned)(size_t)(&svb[0][0]);
  const unsigned voff1 = (unsigned)(size_t)(&svb[1][0]);

#define STAGE_CHUNK(BUF, KT)                                                  \
  do {                                                                        \
    async_ld16(((BUF) ? koff1 : koff0) + (unsigned)(tid * 16),                \
               kbase + (size_t)(KT)*DH_DIM + tid * 8);                        \
    async_ld16(((BUF) ? voff1 : voff0) + (unsigned)((sdh * 32 + spc * 8) * 2),\
               vbase + (size_t)sdh * T_LEN + (KT) + spc * 8);                 \
  } while (0)

  // Stage mask once per block (shared across all 8 waves; same batch).
  const int* mrow = mask + b_ * T_LEN;
  for (int i = tid; i < T_LEN; i += 256) {
    const float m = (float)mrow[i];
    smadd[i] = (1.0f - m) * NEG_BIG;
    smmul[i] = m;
  }

  STAGE_CHUNK(0, 0);                       // prefetch first chunk
  __syncthreads();                         // mask visible to all waves

  const int n   = lane & 15;
  const int lh  = lane >> 4;
  const int klo = lh * 8;       // A-layout interleave base
  const int off = lh * 8;       // C/D layout M offset for this lane half

  const bf16* qbase = q + ((size_t)bh * T_LEN + qt * 16 + n) * DH_DIM;

  // B operands (Q^T), loaded once: table order, dh = half*8 + e (e<8), 16 + half*8 + e-8 (+32).
  const v16bf bq0 = cat(ld8(qbase + lh * 8),      ld8(qbase + lh * 8 + 16));
  const v16bf bq1 = cat(ld8(qbase + 32 + lh * 8), ld8(qbase + 32 + lh * 8 + 16));

  float rmax = -3.0e38f, rsum = 0.f;
  v8f acc0 = zero8(), acc1 = zero8(), acc2 = zero8(), acc3 = zero8();
  bf16* pb = &spb[wave][0];

  for (int kt = 0; kt < T_LEN; kt += 32) {
    const int cur = (kt >> 5) & 1;
    if (kt + 32 < T_LEN) {
      STAGE_CHUNK(cur ^ 1, kt + 32);       // overlap next DMA with compute
      asm volatile("s_wait_asynccnt 0x2" ::: "memory");  // cur chunk landed
    } else {
      asm volatile("s_wait_asynccnt 0x0" ::: "memory");
    }
    __syncthreads();                       // cur chunk visible block-wide

    // ---- S^T = K * Q^T for two 16-key tiles (A operands from LDS) --------
    const bf16* k0 = &skb[cur][0] + (size_t)n * DH_DIM;
    const bf16* k1 = k0 + 16 * DH_DIM;
    v16bf a00 = cat(ld8(k0 + klo),      ld8(k0 + klo + 16));       // dh 0..31
    v16bf a01 = cat(ld8(k0 + 32 + klo), ld8(k0 + 32 + klo + 16));  // dh 32..63
    v16bf a10 = cat(ld8(k1 + klo),      ld8(k1 + klo + 16));
    v16bf a11 = cat(ld8(k1 + 32 + klo), ld8(k1 + 32 + klo + 16));

    v8f s0 = zero8(), s1 = zero8();
    s0 = wmma_bf16(a00, bq0, s0);
    s0 = wmma_bf16(a01, bq1, s0);
    s1 = wmma_bf16(a10, bq0, s1);
    s1 = wmma_bf16(a11, bq1, s1);

    // ---- scale + additive mask; chunk max --------------------------------
    float cmax = -3.0e38f;
#pragma unroll
    for (int r = 0; r < 8; ++r) {
      const float x0 = s0[r] * SM_SCALE - smadd[kt + off + r];
      const float x1 = s1[r] * SM_SCALE - smadd[kt + 16 + off + r];
      s0[r] = x0; s1[r] = x1;
      cmax = fmaxf(cmax, fmaxf(x0, x1));
    }
    cmax = fmaxf(cmax, __shfl_xor(cmax, 16));   // merge the two lane halves
    const float nmax = fmaxf(rmax, cmax);
    const float corr = __expf(rmax - nmax);
    rmax = nmax;
    rsum *= corr;
    acc0 = acc0 * corr; acc1 = acc1 * corr;
    acc2 = acc2 * corr; acc3 = acc3 * corr;

    // ---- p = exp(s - max) * m ; partial row sums -------------------------
    float psum = 0.f;
#pragma unroll
    for (int r = 0; r < 8; ++r) {
      const float p0 = __expf(s0[r] - nmax) * smmul[kt + off + r];
      const float p1 = __expf(s1[r] - nmax) * smmul[kt + 16 + off + r];
      s0[r] = p0; s1[r] = p1;
      psum += p0 + p1;
    }
    rsum += psum;

    // ---- stage P^T in LDS as [query n][key 0..31] bf16 -------------------
#pragma unroll
    for (int r = 0; r < 8; r += 2) {
      v2bf p0 = { (bf16)s0[r], (bf16)s0[r + 1] };
      v2bf p1 = { (bf16)s1[r], (bf16)s1[r + 1] };
      *(v2bf*)(pb + n * 32 + off + r)      = p0;   // keys kt+off+r..+1
      *(v2bf*)(pb + n * 32 + 16 + off + r) = p1;   // keys kt+16+off+r..+1
    }
    asm volatile("s_wait_dscnt 0" ::: "memory");   // same-wave LDS RAW

    // B-layout reload (table order): half h -> keys 8h..8h+7, 16+8h..16+8h+7.
    const bf16* prow = pb + n * 32 + lh * 8;
    const v16bf bp = cat(ld8(prow), ld8(prow + 16));

    // ---- O^T += V^T * P^T for the four 16-dh tiles (A from LDS) ----------
    const bf16* v0 = &svb[cur][0] + (size_t)n * 32;   // dh-tile 0, M = n
    v16bf av0 = cat(ld8(v0 + klo),           ld8(v0 + klo + 16));
    v16bf av1 = cat(ld8(v0 + 16 * 32 + klo), ld8(v0 + 16 * 32 + klo + 16));
    v16bf av2 = cat(ld8(v0 + 32 * 32 + klo), ld8(v0 + 32 * 32 + klo + 16));
    v16bf av3 = cat(ld8(v0 + 48 * 32 + klo), ld8(v0 + 48 * 32 + klo + 16));
    acc0 = wmma_bf16(av0, bp, acc0);
    acc1 = wmma_bf16(av1, bp, acc1);
    acc2 = wmma_bf16(av2, bp, acc2);
    acc3 = wmma_bf16(av3, bp, acc3);

    __syncthreads();        // all waves done with cur before it is re-DMAed
  }

  // ---- finalize: merge lane-half sums, normalize, store ------------------
  rsum += __shfl_xor(rsum, 16);
  const float inv = (rsum > 0.f) ? (1.0f / rsum) : 0.0f;   // all-masked -> 0

  float* orow = out + ((size_t)(b_ * T_LEN + qt * 16 + n)) * D_DIM +
                h * DH_DIM + off;
  const v8f a[4] = {acc0, acc1, acc2, acc3};
#pragma unroll
  for (int dt = 0; dt < 4; ++dt) {
    v4f lo = { a[dt][0] * inv, a[dt][1] * inv, a[dt][2] * inv, a[dt][3] * inv };
    v4f hi = { a[dt][4] * inv, a[dt][5] * inv, a[dt][6] * inv, a[dt][7] * inv };
    *(v4f*)(orow + dt * 16 + 0) = lo;
    *(v4f*)(orow + dt * 16 + 4) = hi;
  }
#undef STAGE_CHUNK
}

// ---------------------------------------------------------------------------
// Host launcher.
// Inputs: x, mask, ln_g, ln_b, wq, bq, wk, bk, wv, bv
// ---------------------------------------------------------------------------
extern "C" void kernel_launch(void* const* d_in, const int* in_sizes, int n_in,
                              void* d_out, int out_size, void* d_ws,
                              size_t ws_size, hipStream_t stream) {
  (void)in_sizes; (void)n_in; (void)out_size; (void)ws_size;

  const float* x    = (const float*)d_in[0];
  const int*   mask = (const int*)d_in[1];
  const float* lng  = (const float*)d_in[2];
  const float* lnb  = (const float*)d_in[3];
  const float* wq   = (const float*)d_in[4];
  const float* bq   = (const float*)d_in[5];
  const float* wk   = (const float*)d_in[6];
  const float* bk   = (const float*)d_in[7];
  const float* wv   = (const float*)d_in[8];
  const float* bv   = (const float*)d_in[9];
  float* out = (float*)d_out;

  // Workspace layout (bf16 elements): Wt(3*512*512) | xn | Q | K | V^T
  bf16* wsb = (bf16*)d_ws;
  bf16* wt  = wsb;
  bf16* xn  = wt + (size_t)3 * 512 * 512;
  bf16* qb  = xn + (size_t)16384 * 512;
  bf16* kb  = qb + (size_t)16384 * 512;
  bf16* vtb = kb + (size_t)16384 * 512;

  prep_weights<<<3 * 512 * 512 / 256, 256, 0, stream>>>(wq, wk, wv, wt);
  ln_kernel<<<16384 / 8, 256, 0, stream>>>(x, lng, lnb, xn);
  proj_kernel<<<dim3(256, 32, 3), 128, 0, stream>>>(xn, wt, bq, bk, bv,
                                                    qb, kb, vtb);
  attn_kernel<<<dim3(8, 128), 256, 0, stream>>>(qb, kb, vtb, mask, out);
}
